// DEE_module_5746666242343
// MI455X (gfx1250) — matrix operand order repaired and checked by reading, not compiled, not run on hardware
//
#include <hip/hip_runtime.h>
#include <hip/hip_bf16.h>
#include <math.h>

// ---------------------------------------------------------------------------
// MI455X / gfx1250 implementation.
// Matrix-heavy stages (3x dilated conv as implicit GEMM, FC1, FC2, 1x1 proj)
// run on v_wmma_f32_16x16x32_f16 (wave32, 16x16 tiles, f32 accumulate).
// Workspace requirement: ~230 MB in d_ws.
// ---------------------------------------------------------------------------

typedef __attribute__((ext_vector_type(16))) _Float16 v16h;
typedef __attribute__((ext_vector_type(8)))  _Float16 v8h;
typedef __attribute__((ext_vector_type(8)))  float    v8f;

#define DEV __device__ __forceinline__

constexpr int B_  = 16;
constexpr int C_  = 256;
constexpr int CQ  = 64;
constexpr int H_  = 64;
constexpr int W_  = 64;
constexpr int HW  = H_ * W_;      // 4096
constexpr int M_  = B_ * HW;      // 65536 GEMM rows
constexpr int MT  = M_ / 16;      // 4096 M tiles
constexpr int HIDc = 170;
constexpr int N1  = 2 * HIDc;     // 340
constexpr int N1P = 352;          // 22 N-tiles
constexpr int K2P = 192;          // 170 padded to 6 K-chunks
constexpr int WF  = 33;           // rfft2 half width
constexpr float TWO_PI_64 = 6.283185307179586f / 64.0f;

// --------------------------- WMMA helpers ----------------------------------

// A-matrix (16x32 f16) lane fragment per ISA layout:
// lanes 0-15 : e=0..7 -> K=base..base+7,   e=8..15 -> K=base+16..base+23
// lanes 16-31: same with base+=8
DEV v16h load_a16(const _Float16* rowptr, int kbase) {
  const v8h* p = (const v8h*)(rowptr + kbase);
  v8h lo = p[0];
  v8h hi = p[2];   // +16 halves
  v16h a;
#pragma unroll
  for (int e = 0; e < 8; ++e) { a[e] = lo[e]; a[e + 8] = hi[e]; }
  return a;
}

DEV v16h zero_a16() {
  v16h a;
#pragma unroll
  for (int e = 0; e < 16; ++e) a[e] = (_Float16)0.0f;
  return a;
}

DEV v8f wmma_f16(v16h a, v16h b, v8f c) {
  return __builtin_amdgcn_wmma_f32_16x16x32_f16(false, a, false, b, (short)0, c,
                                                false, false);
}

// --------------------------- prep kernels ----------------------------------

// x NCHW f32 -> xh NHWC f16  (i indexes NHWC)
__global__ void k_cast_nhwc(const float* __restrict__ x,
                            _Float16* __restrict__ xh) {
  int i = blockIdx.x * blockDim.x + threadIdx.x;
  if (i >= M_ * C_) return;
  int ci = i & 255;
  int r  = i >> 8;               // b*HW + hw
  int hw = r & 4095;
  int b  = r >> 12;
  xh[i] = (_Float16)x[(((size_t)b * C_ + ci) << 12) + hw];
}

// out[:, 0:256] = x
__global__ void k_copy_x(const float* __restrict__ x, float* __restrict__ out) {
  int i = blockIdx.x * blockDim.x + threadIdx.x;
  if (i >= M_ * C_) return;
  int hw = i & 4095;
  int c  = (i >> 12) & 255;
  int b  = i >> 20;
  out[(((size_t)b * 768 + c) << 12) + hw] = x[i];
}

// Pack fp32 K x N matrix into WMMA B fragments (16 contiguous halves/lane).
// B layout: col n = lane%16, K = kc*32 + e + 8*(e>=8) + 8*(lane>=16). Zero pad OOB.
// dst[(nt*kchunks + kc)*512 + lane*16 + e]
__global__ void k_pack_b(const float* __restrict__ Wsrc,
                         _Float16* __restrict__ dst, int ntiles, int kchunks,
                         int Ksrc, int Nsrc, long kStride, long nStride) {
  int i = blockIdx.x * blockDim.x + threadIdx.x;
  int total = ntiles * kchunks * 512;
  if (i >= total) return;
  int e     = i & 15;
  int lane  = (i >> 4) & 31;
  int chunk = i >> 9;
  int kc    = chunk % kchunks;
  int nt    = chunk / kchunks;
  int k = kc * 32 + e + ((e >= 8) ? 8 : 0) + ((lane >= 16) ? 8 : 0);
  int n = nt * 16 + (lane & 15);
  float v = (k < Ksrc && n < Nsrc) ? Wsrc[(long)k * kStride + (long)n * nStride]
                                   : 0.0f;
  dst[i] = (_Float16)v;
}

// ------------------- fused 3-dilation conv (implicit GEMM) -----------------
// y0[b, co, h, w] = (conv_d1 + conv_d2 + conv_d3)/3 via 27-tap WMMA GEMM.
__global__ void k_conv3dil_wmma(const _Float16* __restrict__ xh,
                                const _Float16* __restrict__ Bp,
                                float* __restrict__ y0) {
  int wave = (blockIdx.x * blockDim.x + threadIdx.x) >> 5;
  int lane = threadIdx.x & 31;
  int mt = wave >> 2;     // 4096 M tiles
  int nt = wave & 3;      // 4 N tiles (64 out channels)
  if (mt >= MT) return;
  int row = (mt << 4) + (lane & 15);
  int b  = row >> 12;
  int hw = row & 4095;
  int h  = hw >> 6;
  int w  = hw & 63;
  int ciOff = (lane >= 16) ? 8 : 0;
  v8f acc = {};
#pragma unroll 1
  for (int tap = 0; tap < 27; ++tap) {
    int d  = tap / 9;
    int ky = (tap % 9) / 3;
    int kx = tap % 3;
    int dil = d + 1;
    int hh = h + (ky - 1) * dil;
    int ww = w + (kx - 1) * dil;
    bool inb = (hh >= 0) && (hh < H_) && (ww >= 0) && (ww < W_);
    const _Float16* arow =
        inb ? xh + ((size_t)((b << 12) + (hh << 6) + ww) << 8) : nullptr;
    const _Float16* bbase = Bp + (size_t)tap * 16384 + ((size_t)nt << 12);
    __builtin_prefetch(bbase, 0, 1);   // global_prefetch_b8
#pragma unroll
    for (int kc = 0; kc < 8; ++kc) {
      v16h a = inb ? load_a16(arow, kc * 32 + ciOff) : zero_a16();
      v16h bm = *(const v16h*)(bbase + (kc << 9) + (lane << 4));
      acc = wmma_f16(a, bm, acc);
    }
  }
  int col   = (nt << 4) + (lane & 15);
  int mbase = (mt << 4) + ((lane >= 16) ? 8 : 0);
#pragma unroll
  for (int v = 0; v < 8; ++v) {
    int m   = mbase + v;
    int bb  = m >> 12;
    int hw2 = m & 4095;
    y0[(((size_t)bb * CQ + col) << 12) + hw2] = acc[v] * (1.0f / 3.0f);
  }
}

// ----------------------------- freq_attn -----------------------------------
// forward DFT along W (real input), ortho scale 1/8. out (img, h, kw)
__global__ void k_dft_rows(const float* __restrict__ y0, float* __restrict__ Fr,
                           float* __restrict__ Fi) {
  int i = blockIdx.x * blockDim.x + threadIdx.x;
  if (i >= 1024 * H_ * WF) return;
  int k   = i % WF;
  int t   = i / WF;
  int h   = t & 63;
  int img = t >> 6;
  const float* src = y0 + ((size_t)img << 12) + (h << 6);
  float step = -TWO_PI_64 * (float)k;
  float ar = 0.f, ai = 0.f;
  for (int w = 0; w < W_; ++w) {
    float s, c;
    __sincosf(step * (float)w, &s, &c);
    float v = src[w];
    ar += v * c;
    ai += v * s;
  }
  Fr[i] = ar * 0.125f;
  Fi[i] = ai * 0.125f;
}

// forward DFT along H (complex). in (img,h,kw) -> out (img,kh,kw)
__global__ void k_dft_cols(const float* __restrict__ Fr,
                           const float* __restrict__ Fi, float* __restrict__ Gr,
                           float* __restrict__ Gi) {
  int i = blockIdx.x * blockDim.x + threadIdx.x;
  if (i >= 1024 * H_ * WF) return;
  int kw  = i % WF;
  int t   = i / WF;
  int kh  = t & 63;
  int img = t >> 6;
  float step = TWO_PI_64 * (float)kh;
  float gr = 0.f, gi = 0.f;
  for (int h = 0; h < H_; ++h) {
    float s, c;
    __sincosf(step * (float)h, &s, &c);
    size_t idx = ((size_t)img * H_ + h) * WF + kw;
    float a = Fr[idx], b = Fi[idx];
    gr += a * c + b * s;   // (a+ib) e^{-i theta}
    gi += b * c - a * s;
  }
  Gr[i] = gr * 0.125f;
  Gi[i] = gi * 0.125f;
}

// per (b,ch) mean & max over 64x33, for real and imag
__global__ void k_fa_stats(const float* __restrict__ Gr,
                           const float* __restrict__ Gi, float* __restrict__ mr,
                           float* __restrict__ xr, float* __restrict__ mi,
                           float* __restrict__ xi) {
  int img = blockIdx.x;
  int tid = threadIdx.x;
  const float* gr = Gr + (size_t)img * (H_ * WF);
  const float* gi = Gi + (size_t)img * (H_ * WF);
  float sr = 0.f, si = 0.f, vr = -3.402823466e38f, vi = -3.402823466e38f;
  for (int j = tid; j < H_ * WF; j += blockDim.x) {
    float a = gr[j], b = gi[j];
    sr += a; si += b;
    vr = fmaxf(vr, a); vi = fmaxf(vi, b);
  }
  __shared__ float s0[256], s1[256], s2[256], s3[256];
  s0[tid] = sr; s1[tid] = si; s2[tid] = vr; s3[tid] = vi;
  __syncthreads();
  for (int off = 128; off > 0; off >>= 1) {
    if (tid < off) {
      s0[tid] += s0[tid + off];
      s1[tid] += s1[tid + off];
      s2[tid] = fmaxf(s2[tid], s2[tid + off]);
      s3[tid] = fmaxf(s3[tid], s3[tid + off]);
    }
    __syncthreads();
  }
  if (tid == 0) {
    mr[img] = s0[0] * (1.0f / (H_ * WF));
    mi[img] = s1[0] * (1.0f / (H_ * WF));
    xr[img] = s2[0];
    xi[img] = s3[0];
  }
}

// tiny MLP: cat[mean,max](128) -> 64 -> relu(4) -> sigmoid(64)
__global__ void k_fa_mlp(const float* __restrict__ mean_,
                         const float* __restrict__ max_,
                         const float* __restrict__ crw,
                         const float* __restrict__ crb,
                         const float* __restrict__ m1w,
                         const float* __restrict__ m1b,
                         const float* __restrict__ m2w,
                         const float* __restrict__ m2b,
                         float* __restrict__ wout) {
  int b  = blockIdx.x;
  int ch = threadIdx.x;            // 64 threads
  __shared__ float red[64];
  __shared__ float hh[4];
  const float* mn = mean_ + b * CQ;
  const float* mx = max_ + b * CQ;
  float acc = crb[ch];
  for (int j = 0; j < CQ; ++j) acc += mn[j] * crw[ch * 128 + j];
  for (int j = 0; j < CQ; ++j) acc += mx[j] * crw[ch * 128 + 64 + j];
  red[ch] = acc;
  __syncthreads();
  if (ch < 4) {
    float a = m1b[ch];
    for (int j = 0; j < CQ; ++j) a += red[j] * m1w[j * 4 + ch];
    hh[ch] = fmaxf(a, 0.0f);
  }
  __syncthreads();
  float o = m2b[ch];
  for (int t = 0; t < 4; ++t) o += hh[t] * m2w[t * 64 + ch];
  wout[b * CQ + ch] = 1.0f / (1.0f + __expf(-o));
}

__global__ void k_fa_apply(float* __restrict__ Gr, float* __restrict__ Gi,
                           const float* __restrict__ wr,
                           const float* __restrict__ wi) {
  int i = blockIdx.x * blockDim.x + threadIdx.x;
  if (i >= 1024 * H_ * WF) return;
  int img = i / (H_ * WF);
  Gr[i] *= wr[img];
  Gi[i] *= wi[img];
}

// inverse DFT along H (complex). in (img,kh,kw) -> out (img,h,kw)
__global__ void k_idft_cols(const float* __restrict__ Gr,
                            const float* __restrict__ Gi,
                            float* __restrict__ Fr, float* __restrict__ Fi) {
  int i = blockIdx.x * blockDim.x + threadIdx.x;
  if (i >= 1024 * H_ * WF) return;
  int kw  = i % WF;
  int t   = i / WF;
  int h   = t & 63;
  int img = t >> 6;
  float step = TWO_PI_64 * (float)h;
  float fr = 0.f, fi = 0.f;
  for (int kh = 0; kh < H_; ++kh) {
    float s, c;
    __sincosf(step * (float)kh, &s, &c);
    size_t idx = ((size_t)img * H_ + kh) * WF + kw;
    float a = Gr[idx], b = Gi[idx];
    fr += a * c - b * s;   // (a+ib) e^{+i theta}
    fi += a * s + b * c;
  }
  Fr[i] = fr * 0.125f;
  Fi[i] = fi * 0.125f;
}

// Hermitian inverse DFT along W -> real y2 (NCHW)
__global__ void k_idft_rows(const float* __restrict__ Fr,
                            const float* __restrict__ Fi,
                            float* __restrict__ y2) {
  int i = blockIdx.x * blockDim.x + threadIdx.x;
  if (i >= 1024 * HW) return;
  int w   = i & 63;
  int h   = (i >> 6) & 63;
  int img = i >> 12;
  const float* fr = Fr + ((size_t)img * H_ + h) * WF;
  const float* fi = Fi + ((size_t)img * H_ + h) * WF;
  float acc = fr[0] + fr[32] * ((w & 1) ? -1.0f : 1.0f);
  float step = TWO_PI_64 * (float)w;
  for (int k = 1; k < 32; ++k) {
    float s, c;
    __sincosf(step * (float)k, &s, &c);
    acc += 2.0f * (fr[k] * c - fi[k] * s);
  }
  y2[i] = acc * 0.125f;
}

// ----------------------------- gated block ---------------------------------

__global__ void k_layernorm(const float* __restrict__ y2,
                            const float* __restrict__ gamma,
                            const float* __restrict__ beta,
                            _Float16* __restrict__ out) {
  int m = blockIdx.x * blockDim.x + threadIdx.x;
  if (m >= M_) return;
  int b = m >> 12, hw = m & 4095;
  const float* base = y2 + ((size_t)b << 18) + hw;
  float vals[CQ];
  float mean = 0.f;
#pragma unroll
  for (int c = 0; c < CQ; ++c) { vals[c] = base[(size_t)c << 12]; mean += vals[c]; }
  mean *= (1.0f / CQ);
  float var = 0.f;
#pragma unroll
  for (int c = 0; c < CQ; ++c) { float d = vals[c] - mean; var += d * d; }
  var *= (1.0f / CQ);
  float rs = rsqrtf(var + 1e-5f);
  _Float16* o = out + (size_t)m * CQ;
#pragma unroll
  for (int c = 0; c < CQ; ++c)
    o[c] = (_Float16)((vals[c] - mean) * rs * gamma[c] + beta[c]);
}

// FC1: (65536 x 64) x (64 x 352pad) + bias -> t1 f16 (ld 352)
__global__ void k_gemm_fc1(const _Float16* __restrict__ A,
                           const _Float16* __restrict__ Bp,
                           const float* __restrict__ bias,
                           _Float16* __restrict__ out) {
  int wave = (blockIdx.x * blockDim.x + threadIdx.x) >> 5;
  int lane = threadIdx.x & 31;
  int nt = wave % 22;
  int mt = wave / 22;
  if (mt >= MT) return;
  int row   = (mt << 4) + (lane & 15);
  int ciOff = (lane >= 16) ? 8 : 0;
  v8f acc = {};
#pragma unroll
  for (int kc = 0; kc < 2; ++kc) {
    v16h a  = load_a16(A + (size_t)row * 64, kc * 32 + ciOff);
    v16h bm = *(const v16h*)(Bp + (((size_t)nt * 2 + kc) << 9) + (lane << 4));
    acc = wmma_f16(a, bm, acc);
  }
  int col   = (nt << 4) + (lane & 15);
  float bv  = (col < N1) ? bias[col] : 0.0f;
  int mbase = (mt << 4) + ((lane >= 16) ? 8 : 0);
#pragma unroll
  for (int v = 0; v < 8; ++v)
    out[(size_t)(mbase + v) * N1P + col] = (_Float16)(acc[v] + bv);
}

// depthwise 7x7 on channels 308..339 of t1 -> cbuf (m, 32) f16
__global__ void k_dw7(const _Float16* __restrict__ t1,
                      const float* __restrict__ wgt,
                      const float* __restrict__ bias,
                      _Float16* __restrict__ cb) {
  int i = blockIdx.x * blockDim.x + threadIdx.x;
  if (i >= M_ * 32) return;
  int c = i & 31;
  int m = i >> 5;
  int b = m >> 12, h = (m >> 6) & 63, w = m & 63;
  float acc = bias[c];
  for (int ky = 0; ky < 7; ++ky) {
    int hh = h + ky - 3;
    if (hh < 0 || hh >= H_) continue;
    for (int kx = 0; kx < 7; ++kx) {
      int ww = w + kx - 3;
      if (ww < 0 || ww >= W_) continue;
      size_t sm = (size_t)((b << 12) + (hh << 6) + ww);
      acc += (float)t1[sm * N1P + 308 + c] * wgt[c * 49 + ky * 7 + kx];
    }
  }
  cb[(size_t)m * 32 + c] = (_Float16)acc;
}

// act[m, k<170] = gelu_exact(g) * (k<138 ? i : conv_c); padded to 192
__global__ void k_gate(const _Float16* __restrict__ t1,
                       const _Float16* __restrict__ cb,
                       _Float16* __restrict__ act) {
  int i = blockIdx.x * blockDim.x + threadIdx.x;
  if (i >= M_ * K2P) return;
  int k = i % K2P;
  size_t m = (size_t)(i / K2P);
  float v = 0.0f;
  if (k < HIDc) {
    float g = (float)t1[m * N1P + k];
    float ge = 0.5f * g * (1.0f + erff(g * 0.7071067811865476f));
    float other = (k < 138) ? (float)t1[m * N1P + HIDc + k]
                            : (float)cb[m * 32 + (k - 138)];
    v = ge * other;
  }
  act[i] = (_Float16)v;
}

// FC2: (65536 x 192pad) x (192 x 64) + bias + residual(y2) -> y3 NCHW f32
__global__ void k_gemm_fc2(const _Float16* __restrict__ A,
                           const _Float16* __restrict__ Bp,
                           const float* __restrict__ bias,
                           const float* __restrict__ y2,
                           float* __restrict__ y3) {
  int wave = (blockIdx.x * blockDim.x + threadIdx.x) >> 5;
  int lane = threadIdx.x & 31;
  int nt = wave & 3;
  int mt = wave >> 2;
  if (mt >= MT) return;
  int row   = (mt << 4) + (lane & 15);
  int ciOff = (lane >= 16) ? 8 : 0;
  v8f acc = {};
#pragma unroll
  for (int kc = 0; kc < 6; ++kc) {
    v16h a  = load_a16(A + (size_t)row * K2P, kc * 32 + ciOff);
    v16h bm = *(const v16h*)(Bp + (((size_t)nt * 6 + kc) << 9) + (lane << 4));
    acc = wmma_f16(a, bm, acc);
  }
  int col   = (nt << 4) + (lane & 15);
  float bv  = bias[col];
  int mbase = (mt << 4) + ((lane >= 16) ? 8 : 0);
#pragma unroll
  for (int v = 0; v < 8; ++v) {
    int m   = mbase + v;
    int bb  = m >> 12;
    int hw2 = m & 4095;
    size_t idx = (((size_t)bb * CQ + col) << 12) + hw2;
    y3[idx] = acc[v] + bv + y2[idx];
  }
}

// spatial_tf: x + tanh(BN(dwconv3(x)))
__global__ void k_spatial(const float* __restrict__ y3,
                          const float* __restrict__ wgt,
                          const float* __restrict__ bias,
                          const float* __restrict__ bng,
                          const float* __restrict__ bnb,
                          const float* __restrict__ bnm,
                          const float* __restrict__ bnv,
                          float* __restrict__ y4) {
  int i = blockIdx.x * blockDim.x + threadIdx.x;
  if (i >= B_ * CQ * HW) return;
  int hw = i & 4095;
  int ch = (i >> 12) & 63;
  int b  = i >> 18;
  int h = hw >> 6, w = hw & 63;
  const float* img = y3 + (((size_t)b * CQ + ch) << 12);
  float acc = bias[ch];
  for (int ky = 0; ky < 3; ++ky) {
    int hh = h + ky - 1;
    if (hh < 0 || hh >= H_) continue;
    for (int kx = 0; kx < 3; ++kx) {
      int ww = w + kx - 1;
      if (ww < 0 || ww >= W_) continue;
      acc += img[(hh << 6) + ww] * wgt[ch * 9 + ky * 3 + kx];
    }
  }
  float scale = bng[ch] * rsqrtf(bnv[ch] + 1e-5f);
  float t = (acc - bnm[ch]) * scale + bnb[ch];
  y4[i] = y3[i] + tanhf(t);
}

// relu + cast to NHWC f16 for proj GEMM A
__global__ void k_relu_nhwc(const float* __restrict__ y4,
                            _Float16* __restrict__ a5) {
  int i = blockIdx.x * blockDim.x + threadIdx.x;
  if (i >= M_ * CQ) return;
  int ch = i & 63;
  int m  = i >> 6;
  int b = m >> 12, hw = m & 4095;
  float v = y4[(((size_t)b * CQ + ch) << 12) + hw];
  a5[i] = (_Float16)fmaxf(v, 0.0f);
}

// proj: (65536 x 64) x (64 x 256) + bias -> d_out at channel offset
__global__ void k_gemm_proj(const _Float16* __restrict__ A,
                            const _Float16* __restrict__ Bp,
                            const float* __restrict__ bias,
                            float* __restrict__ out, int chOff) {
  int wave = (blockIdx.x * blockDim.x + threadIdx.x) >> 5;
  int lane = threadIdx.x & 31;
  int nt = wave & 15;
  int mt = wave >> 4;
  if (mt >= MT) return;
  int row   = (mt << 4) + (lane & 15);
  int ciOff = (lane >= 16) ? 8 : 0;
  v8f acc = {};
#pragma unroll
  for (int kc = 0; kc < 2; ++kc) {
    v16h a  = load_a16(A + (size_t)row * 64, kc * 32 + ciOff);
    v16h bm = *(const v16h*)(Bp + (((size_t)nt * 2 + kc) << 9) + (lane << 4));
    acc = wmma_f16(a, bm, acc);
  }
  int col   = (nt << 4) + (lane & 15);
  float bv  = bias[col];
  int mbase = (mt << 4) + ((lane >= 16) ? 8 : 0);
#pragma unroll
  for (int v = 0; v < 8; ++v) {
    int m   = mbase + v;
    int bb  = m >> 12;
    int hw2 = m & 4095;
    out[(((size_t)bb * 768 + chOff + col) << 12) + hw2] = acc[v] + bv;
  }
}

// ---------------------------------------------------------------------------

extern "C" void kernel_launch(void* const* d_in, const int* in_sizes, int n_in,
                              void* d_out, int out_size, void* d_ws,
                              size_t ws_size, hipStream_t stream) {
  (void)in_sizes; (void)n_in; (void)out_size; (void)ws_size;
  const float* x       = (const float*)d_in[0];
  const float* d1_w    = (const float*)d_in[1];
  const float* d2_w    = (const float*)d_in[2];
  const float* d3_w    = (const float*)d_in[3];
  const float* proj_w  = (const float*)d_in[4];
  const float* proj_b  = (const float*)d_in[5];
  const float* fa_cr_w = (const float*)d_in[6];
  const float* fa_cr_b = (const float*)d_in[7];
  const float* fa_m1_w = (const float*)d_in[8];
  const float* fa_m1_b = (const float*)d_in[9];
  const float* fa_m2_w = (const float*)d_in[10];
  const float* fa_m2_b = (const float*)d_in[11];
  const float* g_ln_g  = (const float*)d_in[12];
  const float* g_ln_b  = (const float*)d_in[13];
  const float* g_f1_w  = (const float*)d_in[14];
  const float* g_f1_b  = (const float*)d_in[15];
  const float* g_cv_w  = (const float*)d_in[16];
  const float* g_cv_b  = (const float*)d_in[17];
  const float* g_f2_w  = (const float*)d_in[18];
  const float* g_f2_b  = (const float*)d_in[19];
  const float* st_cv_w = (const float*)d_in[20];
  const float* st_cv_b = (const float*)d_in[21];
  const float* st_bn_g = (const float*)d_in[22];
  const float* st_bn_b = (const float*)d_in[23];
  const float* st_bn_m = (const float*)d_in[24];
  const float* st_bn_v = (const float*)d_in[25];
  float* out = (float*)d_out;

  // ---- workspace carve (offsets aligned to 256 B) ----
  size_t off = 0;
  char* ws = (char*)d_ws;
  auto carve = [&](size_t bytes) -> void* {
    void* p = ws + off;
    off += (bytes + 255) & ~(size_t)255;
    return p;
  };
  _Float16* xh   = (_Float16*)carve((size_t)M_ * C_ * 2);       // 33.5 MB
  _Float16* wpcv = (_Float16*)carve((size_t)27 * 16384 * 2);    // conv B
  _Float16* wp1  = (_Float16*)carve((size_t)22 * 2 * 512 * 2);  // fc1 B
  _Float16* wp2  = (_Float16*)carve((size_t)4 * 6 * 512 * 2);   // fc2 B
  _Float16* wpp  = (_Float16*)carve((size_t)16 * 2 * 512 * 2);  // proj B
  float* y0 = (float*)carve((size_t)B_ * CQ * HW * 4);
  float* Fr = (float*)carve((size_t)1024 * H_ * WF * 4);
  float* Fi = (float*)carve((size_t)1024 * H_ * WF * 4);
  float* Gr = (float*)carve((size_t)1024 * H_ * WF * 4);
  float* Gi = (float*)carve((size_t)1024 * H_ * WF * 4);
  float* mr = (float*)carve((size_t)B_ * CQ * 4);
  float* xr = (float*)carve((size_t)B_ * CQ * 4);
  float* mi = (float*)carve((size_t)B_ * CQ * 4);
  float* xi = (float*)carve((size_t)B_ * CQ * 4);
  float* wr = (float*)carve((size_t)B_ * CQ * 4);
  float* wi = (float*)carve((size_t)B_ * CQ * 4);
  float* y2 = (float*)carve((size_t)B_ * CQ * HW * 4);
  _Float16* tln  = (_Float16*)carve((size_t)M_ * CQ * 2);
  _Float16* t1   = (_Float16*)carve((size_t)M_ * N1P * 2);
  _Float16* cbuf = (_Float16*)carve((size_t)M_ * 32 * 2);
  _Float16* act  = (_Float16*)carve((size_t)M_ * K2P * 2);
  float* y3 = (float*)carve((size_t)B_ * CQ * HW * 4);
  float* y4 = (float*)carve((size_t)B_ * CQ * HW * 4);
  _Float16* a5 = (_Float16*)carve((size_t)M_ * CQ * 2);

  const dim3 T(256);
  auto blk = [](long n) { return dim3((unsigned)((n + 255) / 256)); };

  // ---- shared prep ----
  k_cast_nhwc<<<blk((long)M_ * C_), T, 0, stream>>>(x, xh);
  k_copy_x<<<blk((long)M_ * C_), T, 0, stream>>>(x, out);

  const long nDft = 1024L * H_ * WF;

  for (int br = 0; br < 2; ++br) {
    const float* dws[3] = {d1_w + (size_t)br * CQ * C_ * 9,
                           d2_w + (size_t)br * CQ * C_ * 9,
                           d3_w + (size_t)br * CQ * C_ * 9};
    const float* pjw = proj_w + (size_t)br * C_ * CQ;
    const float* pjb = proj_b + (size_t)br * C_;
    const float* crw = fa_cr_w + (size_t)br * CQ * 128;
    const float* crb = fa_cr_b + (size_t)br * CQ;
    const float* m1w = fa_m1_w + (size_t)br * CQ * 4;
    const float* m1b = fa_m1_b + (size_t)br * 4;
    const float* m2w = fa_m2_w + (size_t)br * 4 * CQ;
    const float* m2b = fa_m2_b + (size_t)br * CQ;
    const float* lng = g_ln_g + (size_t)br * CQ;
    const float* lnb = g_ln_b + (size_t)br * CQ;
    const float* f1w = g_f1_w + (size_t)br * CQ * N1;
    const float* f1b = g_f1_b + (size_t)br * N1;
    const float* cvw = g_cv_w + (size_t)br * 32 * 49;
    const float* cvb = g_cv_b + (size_t)br * 32;
    const float* f2w = g_f2_w + (size_t)br * HIDc * CQ;
    const float* f2b = g_f2_b + (size_t)br * CQ;
    const float* stw = st_cv_w + (size_t)br * CQ * 9;
    const float* stb = st_cv_b + (size_t)br * CQ;
    const float* bng = st_bn_g + (size_t)br * CQ;
    const float* bnb = st_bn_b + (size_t)br * CQ;
    const float* bnm = st_bn_m + (size_t)br * CQ;
    const float* bnv = st_bn_v + (size_t)br * CQ;

    // ---- pack B fragments ----
    for (int tap = 0; tap < 27; ++tap) {
      int d = tap / 9, ky = (tap % 9) / 3, kx = tap % 3;
      const float* Wt = dws[d] + (ky * 3 + kx);
      k_pack_b<<<blk(4L * 8 * 512), T, 0, stream>>>(
          Wt, wpcv + (size_t)tap * 16384, 4, 8, C_, CQ, 9L, (long)C_ * 9);
    }
    k_pack_b<<<blk(22L * 2 * 512), T, 0, stream>>>(f1w, wp1, 22, 2, CQ, N1,
                                                   (long)N1, 1L);
    k_pack_b<<<blk(4L * 6 * 512), T, 0, stream>>>(f2w, wp2, 4, 6, HIDc, CQ,
                                                  (long)CQ, 1L);
    k_pack_b<<<blk(16L * 2 * 512), T, 0, stream>>>(pjw, wpp, 16, 2, CQ, C_, 1L,
                                                   (long)CQ);

    // ---- fused dilated convs (WMMA implicit GEMM) ----
    k_conv3dil_wmma<<<dim3(2048), T, 0, stream>>>(xh, wpcv, y0);

    // ---- freq_attn ----
    k_dft_rows<<<blk(nDft), T, 0, stream>>>(y0, Fr, Fi);
    k_dft_cols<<<blk(nDft), T, 0, stream>>>(Fr, Fi, Gr, Gi);
    k_fa_stats<<<dim3(1024), T, 0, stream>>>(Gr, Gi, mr, xr, mi, xi);
    k_fa_mlp<<<dim3(16), dim3(64), 0, stream>>>(mr, xr, crw, crb, m1w, m1b, m2w,
                                                m2b, wr);
    k_fa_mlp<<<dim3(16), dim3(64), 0, stream>>>(mi, xi, crw, crb, m1w, m1b, m2w,
                                                m2b, wi);
    k_fa_apply<<<blk(nDft), T, 0, stream>>>(Gr, Gi, wr, wi);
    k_idft_cols<<<blk(nDft), T, 0, stream>>>(Gr, Gi, Fr, Fi);
    k_idft_rows<<<blk(1024L * HW), T, 0, stream>>>(Fr, Fi, y2);

    // ---- gated block ----
    k_layernorm<<<blk(M_), T, 0, stream>>>(y2, lng, lnb, tln);
    k_gemm_fc1<<<dim3(11264), T, 0, stream>>>(tln, wp1, f1b, t1);
    k_dw7<<<blk((long)M_ * 32), T, 0, stream>>>(t1, cvw, cvb, cbuf);
    k_gate<<<blk((long)M_ * K2P), T, 0, stream>>>(t1, cbuf, act);
    k_gemm_fc2<<<dim3(2048), T, 0, stream>>>(act, wp2, f2b, y2, y3);

    // ---- spatial_tf + proj ----
    k_spatial<<<blk((long)B_ * CQ * HW), T, 0, stream>>>(y3, stw, stb, bng, bnb,
                                                         bnm, bnv, y4);
    k_relu_nhwc<<<blk((long)M_ * CQ), T, 0, stream>>>(y4, a5);
    k_gemm_proj<<<dim3(8192), T, 0, stream>>>(a5, wpp, pjb, out,
                                              256 * (1 + br));
  }
}
